// ScanCacheAttention_66924180407491
// MI455X (gfx1250) — compile-verified, matrix operand order checked by reading, not run on hardware
//
#include <hip/hip_runtime.h>
#include <hip/hip_bf16.h>
#include <math.h>

typedef __attribute__((ext_vector_type(16))) __bf16 v16bf;
typedef __attribute__((ext_vector_type(8)))  float  v8f;
typedef int vs4i __attribute__((__vector_size__(16)));   // int4, vector_size form

#define L_SEQ   4096
#define EMB_D   2048
#define KVPROJ  512           // KVH * DKQ
#define SLOTS   32
#define NROWS   1024          // 512 key rows + 512 value rows
#define NCHUNK  256
#define CLEN    16            // NCHUNK * CLEN == L_SEQ
#define GPER    1024          // gate tile period
#define EPS_F   1e-6f

// CDNA5 async global->LDS copy (ASYNCcnt path), guarded so an unknown builtin
// name degrades to a synchronous fallback instead of a compile failure.
#if __has_builtin(__builtin_amdgcn_global_load_async_to_lds_b128)
#define HAVE_ASYNC_LDS 1
#else
#define HAVE_ASYNC_LDS 0
#endif

__device__ __forceinline__ void copy16_g2l(const __bf16* g, __bf16* l) {
#if HAVE_ASYNC_LDS
  __builtin_amdgcn_global_load_async_to_lds_b128(
      (__attribute__((address_space(1))) vs4i*)g,
      (__attribute__((address_space(3))) vs4i*)l, 0, 0);
#else
  uint4 p = *(const uint4*)g;
  unsigned* d = (unsigned*)l;
  d[0] = p.x; d[1] = p.y; d[2] = p.z; d[3] = p.w;
#endif
}

template <int NPEND>
__device__ __forceinline__ void wait_async() {
#if HAVE_ASYNC_LDS
#if __has_builtin(__builtin_amdgcn_s_wait_asynccnt)
  __builtin_amdgcn_s_wait_asynccnt((short)NPEND);
#else
  if (NPEND == 0) asm volatile("s_wait_asynccnt 0x0" ::: "memory");
  else            asm volatile("s_wait_asynccnt 0x4" ::: "memory");
#endif
#endif
}

// ---------------------------------------------------------------------------
// f32 -> bf16 conversion with folded scale
// ---------------------------------------------------------------------------
__global__ void f32_to_bf16(const float* __restrict__ in, __bf16* __restrict__ out,
                            int n, float scale) {
  int i = blockIdx.x * blockDim.x + threadIdx.x;
  int stride = gridDim.x * blockDim.x;
  for (; i < n; i += stride) out[i] = (__bf16)(in[i] * scale);
}

// ---------------------------------------------------------------------------
// Tiled transpose + f32->bf16: in [R][Cc] f32 -> out [Cc][R] bf16 (weights,
// done once so both GEMM operands stream contiguous K-major rows).
// ---------------------------------------------------------------------------
__global__ __launch_bounds__(256) void transpose_to_bf16(
    const float* __restrict__ in, __bf16* __restrict__ out,
    int R, int Cc, float scale) {
  __shared__ float tile[32][33];
  int c0 = blockIdx.x * 32, r0 = blockIdx.y * 32;
  int tx = threadIdx.x & 31, ty = threadIdx.x >> 5;    // 32x8
  for (int i = ty; i < 32; i += 8)
    tile[i][tx] = in[(size_t)(r0 + i) * Cc + c0 + tx];
  __syncthreads();
  for (int i = ty; i < 32; i += 8)
    out[(size_t)(c0 + i) * R + r0 + tx] = (__bf16)(tile[tx][i] * scale);
}

// ---------------------------------------------------------------------------
// bf16 WMMA GEMM: C[M,N] = A[M,K] @ BT[N,K]^T, f32 accumulate/output.
// N,K compile-time so epilogue/addressing folds to immediate offsets.
// 128x128 block tile, 32 K-step, 256 threads (8 waves), each wave 32x64.
// Double-buffered LDS tiles fed by async global->LDS b128 copies.
// ---------------------------------------------------------------------------
template <int N, int K>
__global__ __launch_bounds__(256) void gemm_bf16_wmma(
    const __bf16* __restrict__ A, const __bf16* __restrict__ BT,
    float* __restrict__ C, int M)
{
  __shared__ __bf16 As[2][128 * 34];   // pad: 17-dword row stride, coprime w/ 64 banks
  __shared__ __bf16 Bs[2][128 * 34];

  const int tid  = threadIdx.x;
  const int lane = tid & 31;
  const int wid  = tid >> 5;
  const int m0 = blockIdx.y * 128, n0 = blockIdx.x * 128;
  const int wm = (wid & 3) * 32;    // wave M offset within tile
  const int wn = (wid >> 2) * 64;   // wave N offset within tile

  v8f acc[2][4];
#pragma unroll
  for (int mt = 0; mt < 2; ++mt)
#pragma unroll
    for (int nt = 0; nt < 4; ++nt)
#pragma unroll
      for (int i = 0; i < 8; ++i) acc[mt][nt][i] = 0.f;

  // tile loader: each thread moves 4x16B segments per (A,B) tile pair
  const int row0 = tid >> 2, col8 = (tid & 3) * 8;
  const int mr = lane & 15;          // fragment row/col within 16
  const int ks = (lane >> 4) * 8;    // ISA: lanes 16-31 hold K+8 pairs

  auto load_tiles = [&](int k0, int b) {
    copy16_g2l(A  + (size_t)(m0 + row0)      * K + k0 + col8, &As[b][ row0       * 34 + col8]);
    copy16_g2l(A  + (size_t)(m0 + row0 + 64) * K + k0 + col8, &As[b][(row0 + 64) * 34 + col8]);
    copy16_g2l(BT + (size_t)(n0 + row0)      * K + k0 + col8, &Bs[b][ row0       * 34 + col8]);
    copy16_g2l(BT + (size_t)(n0 + row0 + 64) * K + k0 + col8, &Bs[b][(row0 + 64) * 34 + col8]);
  };

  load_tiles(0, 0);
  for (int k0 = 0; k0 < K; k0 += 32) {
    const int cur = (k0 >> 5) & 1;
    if (k0 + 32 < K) { load_tiles(k0 + 32, cur ^ 1); wait_async<4>(); }
    else             { wait_async<0>(); }
    __syncthreads();

    // Fragments per ISA 16-bit A 16x32 layout (pairs (K,K+1) per dword).
    v16bf af[2], bf[4];
#pragma unroll
    for (int mt = 0; mt < 2; ++mt) {
      const __bf16* ap = &As[cur][(size_t)(wm + mt * 16 + mr) * 34];
      v16bf f;
#pragma unroll
      for (int vv = 0; vv < 8; ++vv) {
        int kb = ((vv < 4) ? (vv * 2) : (16 + (vv - 4) * 2)) + ks;
        unsigned u = *(const unsigned*)(ap + kb);
        f[2 * vv]     = ((const __bf16*)&u)[0];
        f[2 * vv + 1] = ((const __bf16*)&u)[1];
      }
      af[mt] = f;
    }
#pragma unroll
    for (int nt = 0; nt < 4; ++nt) {
      const __bf16* bp = &Bs[cur][(size_t)(wn + nt * 16 + mr) * 34];
      v16bf f;
#pragma unroll
      for (int vv = 0; vv < 8; ++vv) {
        int kb = ((vv < 4) ? (vv * 2) : (16 + (vv - 4) * 2)) + ks;
        unsigned u = *(const unsigned*)(bp + kb);
        f[2 * vv]     = ((const __bf16*)&u)[0];
        f[2 * vv + 1] = ((const __bf16*)&u)[1];
      }
      bf[nt] = f;
    }

#pragma unroll
    for (int mt = 0; mt < 2; ++mt)
#pragma unroll
      for (int nt = 0; nt < 4; ++nt)
        acc[mt][nt] = __builtin_amdgcn_wmma_f32_16x16x32_bf16(
            false, af[mt], false, bf[nt], (short)0, acc[mt][nt], false, false);
    __syncthreads();
  }

  // C/D layout: N = lane%16, M = reg + (lane/16)*8 ; N compile-time -> imm offsets
  const int nc = lane & 15, mb = (lane >> 4) * 8;
#pragma unroll
  for (int mt = 0; mt < 2; ++mt)
#pragma unroll
    for (int nt = 0; nt < 4; ++nt) {
      float* cp = C + (size_t)(m0 + wm + mt * 16 + mb) * N + (n0 + wn + nt * 16 + nc);
#pragma unroll
      for (int r = 0; r < 8; ++r) cp[(size_t)r * N] = acc[mt][nt][r];
    }
}

// ---------------------------------------------------------------------------
// Phase 1a: per-chunk gate products P[c] = G_{t0} * G_{t0+1} * ... (left->right)
// ---------------------------------------------------------------------------
__global__ __launch_bounds__(1024) void gate_products(const float* __restrict__ gates,
                                                      float* __restrict__ P) {
  const int c = blockIdx.x;
  const int i = threadIdx.x >> 5, j = threadIdx.x & 31;
  __shared__ float Ms[32][33];
  __shared__ float Gs[32][33];
  const int t0 = c * CLEN;
  Ms[i][j] = gates[(size_t)(t0 & (GPER - 1)) * 1024 + threadIdx.x];
  __syncthreads();
  for (int s = 1; s < CLEN; ++s) {
    int t = t0 + s;
    Gs[i][j] = gates[(size_t)(t & (GPER - 1)) * 1024 + threadIdx.x];
    __syncthreads();
    float acc = 0.f;
#pragma unroll
    for (int kk = 0; kk < 32; ++kk) acc += Ms[i][kk] * Gs[kk][j];
    __syncthreads();
    Ms[i][j] = acc;
    __syncthreads();
  }
  P[(size_t)c * 1024 + threadIdx.x] = Ms[i][j];
}

// Shared scan step: each wave owns rows [rg*32, rg*32+32); lane j owns slot j.
// st rows are wave-private -> wave-lockstep makes read-then-write per row safe.
__device__ __forceinline__ void scan_step(float (*st)[33], float (*Gs)[33],
                                          const float* __restrict__ Kp,
                                          const float* __restrict__ Vp,
                                          int t, int j, int rg) {
  float gcol[32];
#pragma unroll
  for (int i = 0; i < 32; ++i) gcol[i] = Gs[i][j];
#pragma unroll 4
  for (int rr = 0; rr < 32; ++rr) {
    int r = rg * 32 + rr;
    float acc = 0.f;
#pragma unroll
    for (int i = 0; i < 32; ++i) acc += st[r][i] * gcol[i];
    if (j == 0)
      acc += (r < 512) ? Kp[(size_t)t * KVPROJ + r]
                       : Vp[(size_t)t * KVPROJ + (r - 512)];
    st[r][j] = acc;
  }
}

// ---------------------------------------------------------------------------
// Phase 1b: per-chunk local scan with zero init; writes final chunk state F[c]
// ---------------------------------------------------------------------------
__global__ __launch_bounds__(1024) void scan_local(const float* __restrict__ gates,
                                                   const float* __restrict__ Kp,
                                                   const float* __restrict__ Vp,
                                                   float* __restrict__ F) {
  __shared__ float st[NROWS][33];
  __shared__ float Gs[32][33];
  const int tid = threadIdx.x;
  const int j = tid & 31, rg = tid >> 5;
  const int c = blockIdx.x, t0 = c * CLEN;
#pragma unroll
  for (int rr = 0; rr < 32; ++rr) st[rg * 32 + rr][j] = 0.f;
  for (int s = 0; s < CLEN; ++s) {
    int t = t0 + s;
    Gs[rg][j] = gates[(size_t)(t & (GPER - 1)) * 1024 + tid];
    __syncthreads();
    scan_step(st, Gs, Kp, Vp, t, j, rg);
    __syncthreads();
  }
#pragma unroll
  for (int rr = 0; rr < 32; ++rr) {
    int r = rg * 32 + rr;
    F[((size_t)c * NROWS + r) * SLOTS + j] = st[r][j];
  }
}

// ---------------------------------------------------------------------------
// Phase 2: sequential combine over chunks (parallel over rows):
//   Sinit[0]=0 ; Sinit[c+1] = Sinit[c] @ P[c] + F[c]
// ---------------------------------------------------------------------------
__global__ __launch_bounds__(256) void prefix_combine(const float* __restrict__ P,
                                                      const float* __restrict__ F,
                                                      float* __restrict__ Sinit) {
  const int tid = threadIdx.x;
  const int j = tid & 31, rl = tid >> 5;      // 8 local rows per block
  const int r = blockIdx.x * 8 + rl;
  __shared__ float S[8][33];
  __shared__ float Pc[32][33];
  S[rl][j] = 0.f;
  __syncthreads();
  for (int c = 0; c < NCHUNK; ++c) {
    Sinit[((size_t)c * NROWS + r) * SLOTS + j] = S[rl][j];
    for (int e = tid; e < 1024; e += 256) Pc[e >> 5][e & 31] = P[(size_t)c * 1024 + e];
    __syncthreads();
    float acc = 0.f;
#pragma unroll
    for (int i = 0; i < 32; ++i) acc += S[rl][i] * Pc[i][j];
    acc += F[((size_t)c * NROWS + r) * SLOTS + j];
    __syncthreads();
    S[rl][j] = acc;
    __syncthreads();
  }
}

// ---------------------------------------------------------------------------
// Phase 3: re-run scan with correct init, fully fused rmsnorm + GQA attention
// per position; emits attn[t, H*DV] as bf16. Whole 1024x32 state lives in LDS
// (CDNA5: 320KB/WGP makes the fully fused form possible).
// ---------------------------------------------------------------------------
__global__ __launch_bounds__(1024) void scan_attend(
    const float* __restrict__ gates, const float* __restrict__ Kp,
    const float* __restrict__ Vp,    const float* __restrict__ Qp,
    const float* __restrict__ Sinit, const float* __restrict__ key_pos,
    const float* __restrict__ query_pos, const float* __restrict__ ln_k,
    const float* __restrict__ ln_v,  __bf16* __restrict__ attnb)
{
  __shared__ float st[NROWS][33];      // 132 KB state (pad 33: conflict-free)
  __shared__ float Gs[32][33];
  __shared__ float kpos[128][33];
  __shared__ float qs[EMB_D];          // (Qp + query_pos) * ln_k_w, per position
  __shared__ float red[1024];
  __shared__ float red2[1024];
  __shared__ float invK[128];          // rsqrt(mean sq) per (g,e), K side
  __shared__ float invV[128];
  __shared__ float qkbuf[512];
  __shared__ float pbuf[512];
  __shared__ float lnvS[128], lnkS[128], qposS[128];

  const int tid = threadIdx.x;
  const int j = tid & 31, rg = tid >> 5;
  const int c = blockIdx.x, t0 = c * CLEN;

#pragma unroll
  for (int rr = 0; rr < 32; ++rr) {
    int r = rg * 32 + rr;
    st[r][j] = Sinit[((size_t)c * NROWS + r) * SLOTS + j];
  }
  for (int e = tid; e < 128 * 32; e += 1024) kpos[e >> 5][e & 31] = key_pos[e];
  if (tid < 128) { lnvS[tid] = ln_v[tid]; lnkS[tid] = ln_k[tid]; qposS[tid] = query_pos[tid]; }
  __syncthreads();

  for (int s = 0; s < CLEN; ++s) {
    const int t = t0 + s;
    Gs[rg][j] = gates[(size_t)(t & (GPER - 1)) * 1024 + tid];
    for (int e = tid; e < EMB_D; e += 1024) {
      int d = e & 127;
      qs[e] = (Qp[(size_t)t * EMB_D + e] + qposS[d]) * lnkS[d];
    }
    __syncthreads();

    scan_step(st, Gs, Kp, Vp, t, j, rg);
    __syncthreads();

    // rmsnorm sum-of-squares partials over d (128) per (g, slot e)
    {
      int e = tid & 31, grp = tid >> 5;
      int g = grp & 3, seg = grp >> 2;   // seg: 0..7, 16 d's each
      int db = seg * 16;
      float pk = 0.f, pv = 0.f;
#pragma unroll 4
      for (int dd = 0; dd < 16; ++dd) {
        int d = db + dd;
        float xk = st[g * 128 + d][e] + kpos[d][e];
        pk += xk * xk;
        float xv = st[512 + g * 128 + d][e];
        pv += xv * xv;
      }
      red[tid] = pk;
      red2[tid] = pv;
    }
    __syncthreads();
    if (tid < 256) {
      int p = tid & 127;
      float ssum = 0.f;
      if (tid < 128) {
#pragma unroll
        for (int sg = 0; sg < 8; ++sg) ssum += red[p + sg * 128];
        invK[p] = rsqrtf(ssum * (1.0f / 128.0f) + EPS_F);
      } else {
#pragma unroll
        for (int sg = 0; sg < 8; ++sg) ssum += red2[p + sg * 128];
        invV[p] = rsqrtf(ssum * (1.0f / 128.0f) + EPS_F);
      }
    }
    __syncthreads();

    // qk[g,h,e] = inv_k * sum_d q_eff * (sk + kpos)  (2 threads per output)
    {
      int o = tid & 511, half = tid >> 9;
      int e = o & 31, h = (o >> 5) & 3, g = o >> 7;
      int hidx = g * 4 + h;
      int db = half * 64;
      float acc = 0.f;
#pragma unroll 8
      for (int dd = 0; dd < 64; ++dd) {
        int d = db + dd;
        acc += qs[hidx * 128 + d] * (st[g * 128 + d][e] + kpos[d][e]);
      }
      red[tid] = acc;
    }
    __syncthreads();
    if (tid < 512) {
      int e = tid & 31, g = tid >> 7;
      qkbuf[tid] = (red[tid] + red[tid + 512]) * invK[g * 32 + e];
    }
    __syncthreads();
    if (tid < 16) { // softmax over 32 slots per (g,h); fold invV into probs
      const float* row = &qkbuf[tid * 32];
      float mx = row[0];
      for (int e = 1; e < 32; ++e) mx = fmaxf(mx, row[e]);
      float sum = 0.f;
      for (int e = 0; e < 32; ++e) sum += __expf(row[e] - mx);
      float inv = 1.0f / sum;
      int g = tid >> 2;
      for (int e = 0; e < 32; ++e)
        pbuf[tid * 32 + e] = __expf(row[e] - mx) * inv * invV[g * 32 + e];
    }
    __syncthreads();
    // attn[g,h,d] = ln_v[d] * sum_e p * sv ; write bf16 for final WMMA GEMM
    for (int oo = tid; oo < 2048; oo += 1024) {
      int d = oo & 127, h = (oo >> 7) & 3, g = oo >> 9;
      const float* pr = &pbuf[(g * 4 + h) * 32];
      int r = 512 + g * 128 + d;
      float acc = 0.f;
#pragma unroll
      for (int e = 0; e < 32; ++e) acc += pr[e] * st[r][e];
      attnb[(size_t)t * EMB_D + oo] = (__bf16)(acc * lnvS[d]);
    }
    __syncthreads();
  }
}

// ---------------------------------------------------------------------------
extern "C" void kernel_launch(void* const* d_in, const int* in_sizes, int n_in,
                              void* d_out, int out_size, void* d_ws, size_t ws_size,
                              hipStream_t stream) {
  (void)in_sizes; (void)n_in; (void)out_size; (void)ws_size;
  const float* q     = (const float*)d_in[0];
  const float* k     = (const float*)d_in[1];
  const float* v     = (const float*)d_in[2];
  const float* wq    = (const float*)d_in[3];
  const float* wk    = (const float*)d_in[4];
  const float* wv    = (const float*)d_in[5];
  const float* wd    = (const float*)d_in[6];
  const float* keyp  = (const float*)d_in[7];
  const float* qpos  = (const float*)d_in[8];
  const float* lnk   = (const float*)d_in[9];
  const float* lnv   = (const float*)d_in[10];
  const float* gates = (const float*)d_in[11];

  const size_t QKV_N = (size_t)L_SEQ * EMB_D;      // 8.4M
  const size_t WQ_N  = (size_t)EMB_D * EMB_D;      // 4.2M
  const size_t WK_N  = (size_t)EMB_D * KVPROJ;     // 1.05M
  const size_t KV_N  = (size_t)L_SEQ * KVPROJ;     // 2.1M
  const size_t P_N   = (size_t)NCHUNK * 1024;
  const size_t F_N   = (size_t)NCHUNK * NROWS * SLOTS;

  char* ws = (char*)d_ws;
  size_t off = 0;
  auto carve = [&](size_t bytes) -> void* {
    off = (off + 255) & ~(size_t)255;
    void* p = ws + off;
    off += bytes;
    return p;
  };
  __bf16* qb    = (__bf16*)carve(QKV_N * 2);
  __bf16* kb    = (__bf16*)carve(QKV_N * 2);
  __bf16* vb    = (__bf16*)carve(QKV_N * 2);
  __bf16* wqT   = (__bf16*)carve(WQ_N * 2);   // [N=2048][K=2048]
  __bf16* wkT   = (__bf16*)carve(WK_N * 2);   // [N=512][K=2048]
  __bf16* wvT   = (__bf16*)carve(WK_N * 2);
  __bf16* wdT   = (__bf16*)carve(WQ_N * 2);
  __bf16* attnb = (__bf16*)carve(QKV_N * 2);
  float*  Qp    = (float*)carve(QKV_N * 4);
  float*  Kp    = (float*)carve(KV_N * 4);
  float*  Vp    = (float*)carve(KV_N * 4);
  float*  P     = (float*)carve(P_N * 4);
  float*  F     = (float*)carve(F_N * 4);
  float*  Sinit = (float*)carve(F_N * 4);

  const float inv_scale = 1.0f / powf(128.0f, 0.25f); // fold DKQ^0.25 into wq & wk

  f32_to_bf16<<<2048, 256, 0, stream>>>(q, qb, (int)QKV_N, 1.0f);
  f32_to_bf16<<<2048, 256, 0, stream>>>(k, kb, (int)QKV_N, 1.0f);
  f32_to_bf16<<<2048, 256, 0, stream>>>(v, vb, (int)QKV_N, 1.0f);
  // weights: transpose once so GEMM streams contiguous K-major rows for B
  transpose_to_bf16<<<dim3(EMB_D / 32, EMB_D / 32), 256, 0, stream>>>(
      wq, wqT, EMB_D, EMB_D, inv_scale);
  transpose_to_bf16<<<dim3(KVPROJ / 32, EMB_D / 32), 256, 0, stream>>>(
      wk, wkT, EMB_D, KVPROJ, inv_scale);
  transpose_to_bf16<<<dim3(KVPROJ / 32, EMB_D / 32), 256, 0, stream>>>(
      wv, wvT, EMB_D, KVPROJ, 1.0f);
  transpose_to_bf16<<<dim3(EMB_D / 32, EMB_D / 32), 256, 0, stream>>>(
      wd, wdT, EMB_D, EMB_D, 1.0f);

  // Projections (bf16 WMMA, f32 out)
  gemm_bf16_wmma<EMB_D, EMB_D><<<dim3(EMB_D / 128, L_SEQ / 128), 256, 0, stream>>>(
      qb, wqT, Qp, L_SEQ);
  gemm_bf16_wmma<KVPROJ, EMB_D><<<dim3(KVPROJ / 128, L_SEQ / 128), 256, 0, stream>>>(
      kb, wkT, Kp, L_SEQ);
  gemm_bf16_wmma<KVPROJ, EMB_D><<<dim3(KVPROJ / 128, L_SEQ / 128), 256, 0, stream>>>(
      vb, wvT, Vp, L_SEQ);

  // Chunked linear-recurrence scan
  gate_products <<<NCHUNK, 1024, 0, stream>>>(gates, P);
  scan_local    <<<NCHUNK, 1024, 0, stream>>>(gates, Kp, Vp, F);
  prefix_combine<<<NROWS / 8, 256, 0, stream>>>(P, F, Sinit);

  // Fused final scan + rmsnorm + softmax attention -> attn (bf16)
  scan_attend<<<NCHUNK, 1024, 0, stream>>>(gates, Kp, Vp, Qp, Sinit,
                                           keyp, qpos, lnk, lnv, attnb);

  // Output projection: d_out = attn @ wd (f32)
  gemm_bf16_wmma<EMB_D, EMB_D><<<dim3(EMB_D / 128, L_SEQ / 128), 256, 0, stream>>>(
      attnb, wdT, (float*)d_out, L_SEQ);
}
